// ONAN_21053929685020
// MI455X (gfx1250) — matrix operand order repaired and checked by reading, not compiled, not run on hardware
//
#include <hip/hip_runtime.h>
#include <hip/hip_fp16.h>

// ---------------------------------------------------------------------------
// GNN: BatchNorm -> per-node GRU over incoming-neighbor mailboxes -> 2 GEMMs
// MI455X strategy: everything matrix-shaped runs through
// v_wmma_f32_16x16x32_f16 (16x16 tiles of nodes x features), weights staged
// in the 320KB WGP LDS, neighbor features gathered from the L2-resident f16
// feature table directly into WMMA A-fragments.
// ---------------------------------------------------------------------------

typedef __attribute__((ext_vector_type(16))) _Float16 v16h;
typedef __attribute__((ext_vector_type(8)))  float    v8f;

union ABFrag { v16h v; uint4 q[2]; };

#define GNN_D       128
#define GNN_MAXDEG  48
#define WAVES_PB    8

__device__ __forceinline__ float gnn_sigmoid(float x) {
    return 1.0f / (1.0f + __expf(-x));
}

__device__ __forceinline__ v8f gnn_splat8(float x) {
    v8f r;
#pragma unroll
    for (int i = 0; i < 8; ++i) r[i] = x;
    return r;
}

// 16x16 output tile accumulation over K=128 (4 WMMA k-tiles of 32).
// Wp is packed so each (kt, CT, lane) owns 16 contiguous halfs (32B).
template<int NT>
__device__ __forceinline__ v8f gnn_wmma4(const ABFrag* A, const _Float16* Wp,
                                         int CT, int lane, v8f acc) {
#pragma unroll
    for (int kt = 0; kt < 4; ++kt) {
        const v16h* bp = (const v16h*)(Wp + (((kt * NT + CT) * 32 + lane) << 4));
        acc = __builtin_amdgcn_wmma_f32_16x16x32_f16(
            false, A[kt].v, false, *bp, (short)0, acc, false, false);
    }
    return acc;
}

// ---------------------------------------------------------------------------
// Setup kernels
// ---------------------------------------------------------------------------

__global__ void gnn_zinit_k(int* deg, int* cursor, float* sums, float* sumsq, int n) {
    int i = blockIdx.x * blockDim.x + threadIdx.x;
    int stride = gridDim.x * blockDim.x;
    for (int k = i; k < n; k += stride) { deg[k] = 0; cursor[k] = 0; }
    if (i < GNN_D) { sums[i] = 0.f; sumsq[i] = 0.f; }
}

__global__ void gnn_bnstats_k(const float* __restrict__ feat,
                              float* sums, float* sumsq, int n, int rpb) {
    int c = threadIdx.x & (GNN_D - 1);
    int h = threadIdx.x >> 7;   // 0/1: two rows interleaved per column
    int r0 = blockIdx.x * rpb;
    int r1 = r0 + rpb; if (r1 > n) r1 = n;
    float s = 0.f, q = 0.f;
    for (int r = r0 + h; r < r1; r += 2) {
        float v = feat[(size_t)r * GNN_D + c];
        s += v; q += v * v;
    }
    __shared__ float ls[GNN_D], lq[GNN_D];
    if (h == 0) { ls[c] = s; lq[c] = q; }
    __syncthreads();
    if (h == 1) { ls[c] += s; lq[c] += q; }
    __syncthreads();
    if (h == 0) { atomicAdd(&sums[c], ls[c]); atomicAdd(&sumsq[c], lq[c]); }
}

__global__ void gnn_bnfin_k(const float* sums, const float* sumsq,
                            float* stats, int n) {
    int c = threadIdx.x;
    float inv = 1.0f / (float)n;
    float mu  = sums[c] * inv;
    float var = sumsq[c] * inv - mu * mu;
    stats[c]          = mu;
    stats[GNN_D + c]  = rsqrtf(var + 1e-5f);
}

__global__ void gnn_bnapply_k(const float* __restrict__ feat,
                              const float* __restrict__ stats,
                              const float* __restrict__ gamma,
                              const float* __restrict__ beta,
                              _Float16* __restrict__ xh, size_t total) {
    size_t i = (size_t)blockIdx.x * blockDim.x + threadIdx.x;
    size_t stride = (size_t)gridDim.x * blockDim.x;
    for (; i < total; i += stride) {
        int c = (int)(i & (GNN_D - 1));
        float v = (feat[i] - stats[c]) * stats[GNN_D + c] * gamma[c] + beta[c];
        xh[i] = (_Float16)v;
    }
}

// Pack W[nout][128] (row-major, out = A @ W^T) into WMMA B-fragment order:
// idx = ((kt*CT + ct)*32 + lane)*16 + j  holds  W[ct*16 + (lane&15)][kt*32 + ((j>>3)<<4) + ((lane>>4)<<3) + (j&7)]
__global__ void gnn_pack_k(const float* __restrict__ W, _Float16* __restrict__ out,
                           int nout) {
    int CT = nout >> 4;
    int total = nout * GNN_D;
    int i = blockIdx.x * blockDim.x + threadIdx.x;
    int stride = gridDim.x * blockDim.x;
    for (; i < total; i += stride) {
        int j    = i & 15;
        int lane = (i >> 4) & 31;
        int rem  = i >> 9;           // kt*CT + ct
        int ct   = rem % CT;
        int kt   = rem / CT;
        int col  = (ct << 4) + (lane & 15);
        int k    = (kt << 5) + ((j >> 3) << 4) + ((lane >> 4) << 3) + (j & 7);
        out[i] = (_Float16)W[(size_t)col * GNN_D + k];
    }
}

__global__ void gnn_deg_k(const int* __restrict__ dst, int* deg, int e) {
    int i = blockIdx.x * blockDim.x + threadIdx.x;
    int stride = gridDim.x * blockDim.x;
    for (; i < e; i += stride) atomicAdd(&deg[dst[i]], 1);
}

__global__ void gnn_scan_k(const int* __restrict__ deg, int* starts, int n) {
    __shared__ int buf[1024];
    __shared__ int carry;
    if (threadIdx.x == 0) carry = 0;
    __syncthreads();
    for (int base = 0; base < n; base += 1024) {
        int idx = base + threadIdx.x;
        int v = (idx < n) ? deg[idx] : 0;
        buf[threadIdx.x] = v;
        __syncthreads();
        for (int off = 1; off < 1024; off <<= 1) {
            int t = (threadIdx.x >= off) ? buf[threadIdx.x - off] : 0;
            __syncthreads();
            buf[threadIdx.x] += t;
            __syncthreads();
        }
        int incl = buf[threadIdx.x];
        if (idx < n) starts[idx] = carry + incl - v;   // exclusive scan
        int tot = buf[1023];
        __syncthreads();
        if (threadIdx.x == 0) carry += tot;
        __syncthreads();
    }
}

__global__ void gnn_fill_k(const int* __restrict__ src, const int* __restrict__ dst,
                           const int* __restrict__ starts, int* cursor,
                           int* nbr, int e) {
    int i = blockIdx.x * blockDim.x + threadIdx.x;
    int stride = gridDim.x * blockDim.x;
    for (; i < e; i += stride) {
        int d = dst[i];
        int p = atomicAdd(&cursor[d], 1);
        nbr[starts[d] + p] = src[i];
    }
}

// ---------------------------------------------------------------------------
// GRU over ragged mailboxes. One wave per 16-node tile; h kept as 8 f32 C
// fragments in VGPRs (+ f16 shadow in a private LDS strip as next A operand).
// Weights (f16 B-fragment order) + biases live in LDS for the whole block.
// ---------------------------------------------------------------------------
__global__ void __launch_bounds__(256)
gnn_gru_k(const _Float16* __restrict__ xh, const int* __restrict__ nbr,
          const int* __restrict__ starts, const int* __restrict__ deg,
          const _Float16* __restrict__ wihp, const _Float16* __restrict__ whhp,
          const float* __restrict__ b_ih, const float* __restrict__ b_hh,
          _Float16* __restrict__ hn, int n) {
    extern __shared__ char smem[];
    _Float16* s_wih = (_Float16*)smem;                 // 384*128 halfs (96KB)
    _Float16* s_whh = s_wih + 384 * GNN_D;             // 96KB
    float*    s_bih = (float*)(s_whh + 384 * GNN_D);   // 384 f32
    float*    s_bhh = s_bih + 384;
    _Float16* s_h   = (_Float16*)(s_bhh + 384);        // 8 waves * 16*128 halfs

    int tid = threadIdx.x;
    {   // cooperative weight staging into LDS
        const uint4* s0 = (const uint4*)wihp; uint4* d0 = (uint4*)s_wih;
        for (int i = tid; i < (384 * GNN_D * 2) / 16; i += 256) d0[i] = s0[i];
        const uint4* s1 = (const uint4*)whhp; uint4* d1 = (uint4*)s_whh;
        for (int i = tid; i < (384 * GNN_D * 2) / 16; i += 256) d1[i] = s1[i];
        for (int i = tid; i < 384; i += 256) { s_bih[i] = b_ih[i]; s_bhh[i] = b_hh[i]; }
    }
    __syncthreads();

    int wave = tid >> 5, lane = tid & 31;
    int lane_r = lane & 15, lane_hi = lane >> 4;
    int tile = blockIdx.x * WAVES_PB + wave;
    int ntiles = (n + 15) >> 4;
    _Float16* hbuf = s_h + wave * (16 * GNN_D);

    {   // zero this wave's h shadow
        uint4 z; z.x = z.y = z.z = z.w = 0u;
        uint4* hb = (uint4*)hbuf;
        for (int i = lane; i < 256; i += 32) hb[i] = z;
    }

    v8f hfrag[8];
#pragma unroll
    for (int ct = 0; ct < 8; ++ct) hfrag[ct] = gnn_splat8(0.f);

    if (tile < ntiles) {
        int node0 = tile << 4;
        int rowNode = node0 + lane_r;
        int rowStart = 0, rowLen = 0;
        if (rowNode < n) {
            rowStart = starts[rowNode];
            int dg = deg[rowNode];
            rowLen = dg < GNN_MAXDEG ? dg : GNN_MAXDEG;
        }
        int lenM[8];
#pragma unroll
        for (int v = 0; v < 8; ++v) {
            int nd = node0 + v + (lane_hi << 3);
            int dg = (nd < n) ? deg[nd] : 0;
            lenM[v] = dg < GNN_MAXDEG ? dg : GNN_MAXDEG;
        }
        int maxlen = 0;   // wave-uniform: every lane reads the same 16 degrees
        for (int i = 0; i < 16; ++i) {
            int nd = node0 + i;
            int dg = (nd < n) ? deg[nd] : 0;
            dg = dg < GNN_MAXDEG ? dg : GNN_MAXDEG;
            maxlen = dg > maxlen ? dg : maxlen;
        }

        for (int t = 0; t < maxlen; ++t) {
            ABFrag mA[4], hA[4];
            bool has = (t < rowLen);
            int srow = has ? nbr[rowStart + t] : 0;
            const uint4* xr = (const uint4*)(xh + (size_t)srow * GNN_D);
            const uint4* hr = (const uint4*)(hbuf + lane_r * GNN_D);
            uint4 z; z.x = z.y = z.z = z.w = 0u;
#pragma unroll
            for (int kt = 0; kt < 4; ++kt) {
                if (has) {
                    mA[kt].q[0] = xr[kt * 4 + lane_hi];
                    mA[kt].q[1] = xr[kt * 4 + 2 + lane_hi];
                } else {
                    mA[kt].q[0] = z; mA[kt].q[1] = z;
                }
                hA[kt].q[0] = hr[kt * 4 + lane_hi];
                hA[kt].q[1] = hr[kt * 4 + 2 + lane_hi];
            }

            for (int ct = 0; ct < 8; ++ct) {
                int col = (ct << 4) + lane_r;
                // r gate
                v8f a0 = gnn_splat8(s_bih[col]);
                v8f a1 = gnn_splat8(s_bhh[col]);
                a0 = gnn_wmma4<24>(mA, s_wih, ct, lane, a0);
                a1 = gnn_wmma4<24>(hA, s_whh, ct, lane, a1);
                v8f rg;
#pragma unroll
                for (int v = 0; v < 8; ++v) rg[v] = gnn_sigmoid(a0[v] + a1[v]);
                // z gate
                a0 = gnn_splat8(s_bih[GNN_D + col]);
                a1 = gnn_splat8(s_bhh[GNN_D + col]);
                a0 = gnn_wmma4<24>(mA, s_wih, 8 + ct, lane, a0);
                a1 = gnn_wmma4<24>(hA, s_whh, 8 + ct, lane, a1);
                v8f zg;
#pragma unroll
                for (int v = 0; v < 8; ++v) zg[v] = gnn_sigmoid(a0[v] + a1[v]);
                // n gate + state update (masked per row by mailbox length)
                a0 = gnn_splat8(s_bih[2 * GNN_D + col]);
                a1 = gnn_splat8(s_bhh[2 * GNN_D + col]);
                a0 = gnn_wmma4<24>(mA, s_wih, 16 + ct, lane, a0);
                a1 = gnn_wmma4<24>(hA, s_whh, 16 + ct, lane, a1);
                v8f h = hfrag[ct];
#pragma unroll
                for (int v = 0; v < 8; ++v) {
                    float nn = tanhf(a0[v] + rg[v] * a1[v]);
                    float hnew = (1.f - zg[v]) * nn + zg[v] * h[v];
                    h[v] = (t < lenM[v]) ? hnew : h[v];
                }
                hfrag[ct] = h;
#pragma unroll
                for (int v = 0; v < 8; ++v)
                    hbuf[(v + (lane_hi << 3)) * GNN_D + col] = (_Float16)h[v];
            }
        }

        // final hidden (f16 shadow is row-major) -> global hn
        uint4* dsth = (uint4*)(hn + (size_t)node0 * GNN_D);
        const uint4* srch = (const uint4*)hbuf;
        for (int i = lane; i < 256; i += 32)
            if (node0 + (i >> 4) < n) dsth[i] = srch[i];
    }
}

// ---------------------------------------------------------------------------
// out = x @ W_self^T + hn @ W_neigh^T  (both accumulated into one C fragment)
// ---------------------------------------------------------------------------
__global__ void __launch_bounds__(256)
gnn_out_k(const _Float16* __restrict__ xh, const _Float16* __restrict__ hnh,
          const _Float16* __restrict__ wsp, const _Float16* __restrict__ wnp,
          float* __restrict__ out, int n) {
    extern __shared__ char smem[];
    _Float16* s_ws = (_Float16*)smem;              // 128*128 halfs (32KB)
    _Float16* s_wn = s_ws + GNN_D * GNN_D;         // 32KB
    int tid = threadIdx.x;
    {
        const uint4* a = (const uint4*)wsp; uint4* b = (uint4*)s_ws;
        for (int i = tid; i < (GNN_D * GNN_D * 2) / 16; i += 256) b[i] = a[i];
        const uint4* c = (const uint4*)wnp; uint4* d = (uint4*)s_wn;
        for (int i = tid; i < (GNN_D * GNN_D * 2) / 16; i += 256) d[i] = c[i];
    }
    __syncthreads();

    int wave = tid >> 5, lane = tid & 31;
    int lane_r = lane & 15, lane_hi = lane >> 4;
    int tile = blockIdx.x * WAVES_PB + wave;
    int ntiles = (n + 15) >> 4;
    if (tile >= ntiles) return;
    int node0 = tile << 4;
    int row = node0 + lane_r;
    bool okrow = row < n;

    ABFrag xA[4], hA[4];
    uint4 z; z.x = z.y = z.z = z.w = 0u;
    const uint4* xr = (const uint4*)(xh + (size_t)row * GNN_D);
    const uint4* hr = (const uint4*)(hnh + (size_t)row * GNN_D);
#pragma unroll
    for (int kt = 0; kt < 4; ++kt) {
        if (okrow) {
            xA[kt].q[0] = xr[kt * 4 + lane_hi];
            xA[kt].q[1] = xr[kt * 4 + 2 + lane_hi];
            hA[kt].q[0] = hr[kt * 4 + lane_hi];
            hA[kt].q[1] = hr[kt * 4 + 2 + lane_hi];
        } else {
            xA[kt].q[0] = z; xA[kt].q[1] = z;
            hA[kt].q[0] = z; hA[kt].q[1] = z;
        }
    }

    for (int ct = 0; ct < 8; ++ct) {
        v8f acc = gnn_splat8(0.f);
        acc = gnn_wmma4<8>(xA, s_ws, ct, lane, acc);
        acc = gnn_wmma4<8>(hA, s_wn, ct, lane, acc);
#pragma unroll
        for (int v = 0; v < 8; ++v) {
            int r = node0 + v + (lane_hi << 3);
            if (r < n) out[(size_t)r * GNN_D + (ct << 4) + lane_r] = acc[v];
        }
    }
}

// ---------------------------------------------------------------------------
// Launch
// ---------------------------------------------------------------------------
extern "C" void kernel_launch(void* const* d_in, const int* in_sizes, int n_in,
                              void* d_out, int out_size, void* d_ws, size_t ws_size,
                              hipStream_t stream) {
    const float* feat   = (const float*)d_in[0];
    const int*   src    = (const int*)d_in[1];
    const int*   dst    = (const int*)d_in[2];
    const float* gamma  = (const float*)d_in[3];
    const float* beta   = (const float*)d_in[4];
    const float* W_ih   = (const float*)d_in[5];
    const float* W_hh   = (const float*)d_in[6];
    const float* b_ih   = (const float*)d_in[7];
    const float* b_hh   = (const float*)d_in[8];
    const float* W_self = (const float*)d_in[9];
    const float* W_neigh= (const float*)d_in[10];
    float* out = (float*)d_out;

    int n = in_sizes[0] / GNN_D;
    int e = in_sizes[1];
    (void)n_in; (void)out_size; (void)ws_size;

    char* ws = (char*)d_ws;
    size_t off = 0;
    auto alloc = [&](size_t bytes) -> char* {
        char* p = ws + off;
        off = (off + bytes + 255) & ~(size_t)255;
        return p;
    };
    float*    sums   = (float*)alloc(GNN_D * 4);
    float*    sumsq  = (float*)alloc(GNN_D * 4);
    float*    stats  = (float*)alloc(2 * GNN_D * 4);
    int*      deg    = (int*)alloc((size_t)n * 4);
    int*      starts = (int*)alloc((size_t)n * 4);
    int*      cursor = (int*)alloc((size_t)n * 4);
    int*      nbr    = (int*)alloc((size_t)e * 4);
    _Float16* xhp    = (_Float16*)alloc((size_t)n * GNN_D * 2);
    _Float16* hnp    = (_Float16*)alloc((size_t)n * GNN_D * 2);
    _Float16* wihp   = (_Float16*)alloc((size_t)384 * GNN_D * 2);
    _Float16* whhp   = (_Float16*)alloc((size_t)384 * GNN_D * 2);
    _Float16* wsp    = (_Float16*)alloc((size_t)GNN_D * GNN_D * 2);
    _Float16* wnp    = (_Float16*)alloc((size_t)GNN_D * GNN_D * 2);

    gnn_zinit_k<<<256, 256, 0, stream>>>(deg, cursor, sums, sumsq, n);

    int rpb = 512;
    gnn_bnstats_k<<<(n + rpb - 1) / rpb, 256, 0, stream>>>(feat, sums, sumsq, n, rpb);
    gnn_bnfin_k<<<1, 128, 0, stream>>>(sums, sumsq, stats, n);
    gnn_bnapply_k<<<1024, 256, 0, stream>>>(feat, stats, gamma, beta, xhp,
                                            (size_t)n * GNN_D);

    gnn_pack_k<<<(384 * GNN_D + 255) / 256, 256, 0, stream>>>(W_ih, wihp, 384);
    gnn_pack_k<<<(384 * GNN_D + 255) / 256, 256, 0, stream>>>(W_hh, whhp, 384);
    gnn_pack_k<<<(GNN_D * GNN_D + 255) / 256, 256, 0, stream>>>(W_self, wsp, GNN_D);
    gnn_pack_k<<<(GNN_D * GNN_D + 255) / 256, 256, 0, stream>>>(W_neigh, wnp, GNN_D);

    gnn_deg_k<<<(e + 255) / 256, 256, 0, stream>>>(dst, deg, e);
    gnn_scan_k<<<1, 1024, 0, stream>>>(deg, starts, n);
    gnn_fill_k<<<(e + 255) / 256, 256, 0, stream>>>(src, dst, starts, cursor, nbr, e);

    int ntiles = (n + 15) / 16;
    int gblocks = (ntiles + WAVES_PB - 1) / WAVES_PB;

    size_t gsm = (size_t)384 * GNN_D * 2 * 2     // W_ih + W_hh (f16)
               + (size_t)384 * 4 * 2             // biases (f32)
               + (size_t)WAVES_PB * 16 * GNN_D * 2; // per-wave h shadow (f16)
    (void)hipFuncSetAttribute((const void*)gnn_gru_k,
                              hipFuncAttributeMaxDynamicSharedMemorySize, (int)gsm);
    gnn_gru_k<<<gblocks, 256, gsm, stream>>>(xhp, nbr, starts, deg,
                                             wihp, whhp, b_ih, b_hh, hnp, n);

    size_t osm = (size_t)GNN_D * GNN_D * 2 * 2;  // W_self + W_neigh (f16)
    (void)hipFuncSetAttribute((const void*)gnn_out_k,
                              hipFuncAttributeMaxDynamicSharedMemorySize, (int)osm);
    gnn_out_k<<<gblocks, 256, osm, stream>>>(xhp, hnp, wsp, wnp, out, n);
}
